// MultiHeadAttention_51144470561199
// MI455X (gfx1250) — matrix operand order repaired and checked by reading, not compiled, not run on hardware
//
#include <hip/hip_runtime.h>

typedef __attribute__((ext_vector_type(16))) __bf16 v16bf;
typedef __attribute__((ext_vector_type(8)))  float  v8f;
typedef unsigned int  u32;
typedef unsigned short u16;

#define BSZ 4
#define TT  2048
#define CC  1024
#define HH  16
#define DH  64
#define MROWS (BSZ*TT)   /* 8192 */
#define NQKV  (3*CC)     /* 3072 */

union Frag32 { uint4 u[2]; v16bf f; };

// load one 32-byte WMMA operand fragment (lane-contiguous) from LDS: 2x ds_load_b128
__device__ __forceinline__ v16bf frag_from_lds(const u32* laneBase) {
  Frag32 fr;
  const uint4* p = (const uint4*)laneBase;
  fr.u[0] = p[0];
  fr.u[1] = p[1];
  return fr.f;
}

// ---------------------------------------------------------------- convert f32 -> bf16 (x4)
__global__ void cvt_f32_bf16_x4(const float4* __restrict__ s, ushort4* __restrict__ d, int n4) {
  int i = blockIdx.x * blockDim.x + threadIdx.x;
  int stride = gridDim.x * blockDim.x;
  for (; i < n4; i += stride) {
    float4 v = s[i];
    ushort4 u;
    u.x = __builtin_bit_cast(u16, (__bf16)v.x);
    u.y = __builtin_bit_cast(u16, (__bf16)v.y);
    u.z = __builtin_bit_cast(u16, (__bf16)v.z);
    u.w = __builtin_bit_cast(u16, (__bf16)v.w);
    d[i] = u;
  }
}

// ---------------------------------------------------------------- generic bf16 GEMM core
// block tile 128(M) x 128(N), K-step 32; 8 waves in 4(M)x2(N) grid of 32x64 wave tiles.
// Software-pipelined: tile kk+32 is fetched to registers while tile kk computes.
// LDS tiles in WMMA fragment order:
//   Afrag[u32]: [8 mtile][32 lane][8 slot]   (A-layout, all-u32 staging)
//   Bfrag[u32]: [8 ntile][32 lane][8 slot]   (B-layout, u16 scatter staging)
#define GEMM_CORE(LDB_)                                                                   \
  const int tid = threadIdx.x, lane = tid & 31, wave = tid >> 5;                          \
  const int wm = wave & 3, wn = wave >> 2;                                                \
  const int mTile = blockIdx.y * 128, nTile = blockIdx.x * 128;                           \
  __shared__ __align__(32) u32 Afrag[8 * 32 * 8];                                         \
  __shared__ __align__(32) u32 Bfrag[8 * 32 * 8];                                         \
  u16* BfragH = (u16*)Bfrag;                                                              \
  v8f acc[2][4] = {};                                                                     \
  const u32* Ag = (const u32*)Abf;                                                        \
  const u32* Bg = (const u32*)Wbf;                                                        \
  u32 aReg[8], bReg[8];                                                                   \
  _Pragma("unroll")                                                                       \
  for (int i = 0; i < 8; ++i) { /* prefetch tile kk=0 */                                  \
    int idx = tid + i * 256;                                                              \
    aReg[i] = Ag[((mTile + (idx >> 4)) * CC) >> 1 | (idx & 15)];                          \
    bReg[i] = Bg[((idx >> 6) * (LDB_) + nTile) >> 1 | (idx & 63)];                        \
  }                                                                                       \
  for (int kk = 0; kk < CC; kk += 32) {                                                   \
    __syncthreads();                                                                      \
    _Pragma("unroll")                                                                     \
    for (int i = 0; i < 8; ++i) { /* A scatter: u32 -> A-layout */                        \
      int idx = tid + i * 256, row = idx >> 4, c2 = idx & 15;                             \
      int laneT = (row & 15) + 16 * ((c2 >> 2) & 1);                                      \
      int slot = (c2 & 3) + 4 * (c2 >> 3);                                                \
      Afrag[(((row >> 4) * 32 + laneT) << 3) + slot] = aReg[i];                           \
    }                                                                                     \
    _Pragma("unroll")                                                                     \
    for (int i = 0; i < 8; ++i) { /* B scatter: 2x u16 -> B-layout */                     \
      int idx = tid + i * 256, k = idx >> 6, c2 = idx & 63;                               \
      int n0 = c2 * 2;                                                                    \
      int kl = 16 * ((k >> 3) & 1), elem = (k & 7) + 8 * ((k >> 4) & 1);                  \
      BfragH[(((n0 >> 4) * 32 + (n0 & 15) + kl) << 4) + elem] = (u16)bReg[i];             \
      BfragH[((((n0 + 1) >> 4) * 32 + ((n0 + 1) & 15) + kl) << 4) + elem] =               \
          (u16)(bReg[i] >> 16);                                                           \
    }                                                                                     \
    __syncthreads();                                                                      \
    if (kk + 32 < CC) { /* prefetch next tile; overlaps WMMA phase */                     \
      _Pragma("unroll")                                                                   \
      for (int i = 0; i < 8; ++i) {                                                       \
        int idx = tid + i * 256;                                                          \
        aReg[i] = Ag[((mTile + (idx >> 4)) * CC + kk + 32) >> 1 | (idx & 15)];            \
        bReg[i] = Bg[((kk + 32 + (idx >> 6)) * (LDB_) + nTile) >> 1 | (idx & 63)];        \
      }                                                                                   \
    }                                                                                     \
    v16bf a[2];                                                                           \
    _Pragma("unroll")                                                                     \
    for (int mi = 0; mi < 2; ++mi)                                                        \
      a[mi] = frag_from_lds(&Afrag[(((wm * 2 + mi) * 32 + lane) << 3)]);                  \
    _Pragma("unroll")                                                                     \
    for (int ni = 0; ni < 4; ++ni) {                                                      \
      v16bf b = frag_from_lds(&Bfrag[(((wn * 4 + ni) * 32 + lane) << 3)]);                \
      _Pragma("unroll")                                                                   \
      for (int mi = 0; mi < 2; ++mi)                                                      \
        acc[mi][ni] = __builtin_amdgcn_wmma_f32_16x16x32_bf16(false, a[mi], false, b,     \
                                                              (short)0, acc[mi][ni],      \
                                                              false, false);              \
    }                                                                                     \
  }

// ---------------------------------------------------------------- QKV GEMM
__launch_bounds__(256)
__global__ void qkv_gemm_wmma(const u16* __restrict__ Abf, const u16* __restrict__ Wbf,
                              const float* __restrict__ bias,
                              u16* __restrict__ Qo, u16* __restrict__ Ko, u16* __restrict__ Vo) {
  GEMM_CORE(NQKV)
  const int rhalf = (lane >> 4) * 8, colL = lane & 15;
#pragma unroll
  for (int ni = 0; ni < 4; ++ni) {
    const int nb = nTile + wn * 64 + ni * 16;       // 16-wide tile: uniform Q/K/V segment
    const int which = nb >> 10, cbase = nb & 1023;
    const int hh = cbase >> 6, dbase = cbase & 63;  // tile stays within one head
    const float bv = bias[nb + colL];
    u16* dstp = (which == 0) ? Qo : (which == 1) ? Ko : Vo;
    const float sc = (which == 0) ? 0.125f : 1.0f;  // fold 1/sqrt(64) into Q
#pragma unroll
    for (int mi = 0; mi < 2; ++mi) {
      const int m0 = mTile + wm * 32 + mi * 16 + rhalf;
      const int bb = m0 >> 11, t0 = m0 & 2047;
      __bf16* dp = (__bf16*)dstp +
                   ((((long long)(bb * HH + hh)) * TT + t0) * DH + dbase + colL);
#pragma unroll
      for (int r = 0; r < 8; ++r)
        dp[r * DH] = (__bf16)((acc[mi][ni][r] + bv) * sc);
    }
  }
}

// ---------------------------------------------------------------- output projection GEMM
__launch_bounds__(256)
__global__ void proj_gemm_wmma(const u16* __restrict__ Abf, const u16* __restrict__ Wbf,
                               const float* __restrict__ bias, float* __restrict__ out) {
  GEMM_CORE(CC)
  const int rhalf = (lane >> 4) * 8, colL = lane & 15;
#pragma unroll
  for (int ni = 0; ni < 4; ++ni) {
    const int n = nTile + wn * 64 + ni * 16 + colL;
    const float bv = bias[n];
#pragma unroll
    for (int mi = 0; mi < 2; ++mi) {
      const int m0 = mTile + wm * 32 + mi * 16 + rhalf;
      float* op = out + (long long)m0 * CC + n;
#pragma unroll
      for (int r = 0; r < 8; ++r)
        op[r * CC] = acc[mi][ni][r] + bv;
    }
  }
}

// ---------------------------------------------------------------- flash attention
// grid (T/128, H, B); 256 threads; wave owns 16 query rows; 64-key blocks, causal.
// Software-pipelined K/V staging. LDS fragment layouts:
//   Kfrag[u32]: [4 keytile][2 dstep][32 lane][8 slot]  (B-layout for S=Q*K^T, u32 staging)
//   Vfrag[u32]: [4 dtile ][2 kstep][32 lane][8 slot]   (B-layout for O=P*V, u16 scatter)
//   Pfrag[u32]: [8 wave][2 kstep][32 lane][8 slot]     (A-layout, written from C-layout)
__launch_bounds__(256)
__global__ void attn_wmma(const u16* __restrict__ Qb, const u16* __restrict__ Kb,
                          const u16* __restrict__ Vb, const int* __restrict__ amask,
                          u16* __restrict__ ctxo) {
  const __bf16* Qg = (const __bf16*)Qb;
  __bf16* ctx = (__bf16*)ctxo;
  __shared__ __align__(32) u32 Kfrag[4 * 2 * 32 * 8];
  __shared__ __align__(32) u32 Vfrag[4 * 2 * 32 * 8];
  __shared__ __align__(32) u32 Pfrag[8 * 2 * 32 * 8];
  __shared__ float mAddS[64];
  u16* VfragH = (u16*)Vfrag;
  u16* PfragH = (u16*)Pfrag;

  const int tid = threadIdx.x, lane = tid & 31, wave = tid >> 5;
  const int qb = blockIdx.x, h = blockIdx.y, b = blockIdx.z;
  const long long headOff = ((long long)(b * HH + h)) * TT * DH;
  const int q0 = qb * 128 + wave * 16;
  const int rhalf = (lane >> 4) * 8, colL = lane & 15;

  // Q fragments resident (Q pre-scaled by 1/8); direct global_load_b128 gathers
  v16bf aQ[2];
  {
    const int koff = (lane >> 4) * 8;  // 0 or 8
    const uint4* qp4 = (const uint4*)(Qg + headOff + (long long)(q0 + colL) * DH);
#pragma unroll
    for (int ks = 0; ks < 2; ++ks) {
      Frag32 fr;
      fr.u[0] = qp4[ks * 4 + (koff >> 2)];
      fr.u[1] = qp4[ks * 4 + (koff >> 2) + 1];
      aQ[ks] = fr.f;
    }
  }
  v8f o[4] = {};
  float mrow[8], lrow[8];
#pragma unroll
  for (int r = 0; r < 8; ++r) { mrow[r] = -1e30f; lrow[r] = 0.f; }

  const u32* KgBase = (const u32*)(Kb) + headOff / 2;
  const u32* VgBase = (const u32*)(Vb) + headOff / 2;

  // prefetch K/V block 0 into registers
  u32 kReg[8], vReg[8];
#pragma unroll
  for (int i = 0; i < 8; ++i) {
    int idx = tid + i * 256, row = idx >> 5, c2 = idx & 31;
    kReg[i] = KgBase[row * 32 + c2];
    vReg[i] = VgBase[row * 32 + c2];
  }

  const int kbMax = qb * 2 + 2;
  for (int kb = 0; kb < kbMax; ++kb) {
    const int kStart = kb * 64;
    __syncthreads();
#pragma unroll
    for (int i = 0; i < 8; ++i) {               // scatter staged K,V regs to LDS frags
      int idx = tid + i * 256, row = idx >> 5, c2 = idx & 31;  // row=key, d=2*c2
      // K -> B-layout for S (cols=keys, K-dim=d): u32 keeps d-pair together
      {
        int nt = row >> 4, laneT = (row & 15) + 16 * ((c2 >> 3) & 1);
        int ks = c2 >> 4, slot = c2 & 7;
        Kfrag[(((nt * 2 + ks) * 32 + laneT) << 3) + slot] = kReg[i];
      }
      // V -> B-layout for PV (cols=d, K-dim=key): scatter two u16
      {
        int ks = row >> 5, elem = (row & 7) + 8 * ((row >> 4) & 1), kl = 16 * ((row >> 3) & 1);
        int d0 = c2 * 2;
        VfragH[((((d0 >> 4) * 2 + ks) * 32 + (d0 & 15) + kl) << 4) + elem] = (u16)vReg[i];
        VfragH[(((((d0 + 1) >> 4) * 2 + ks) * 32 + ((d0 + 1) & 15) + kl) << 4) + elem] =
            (u16)(vReg[i] >> 16);
      }
    }
    if (tid < 64) mAddS[tid] = amask[b * TT + kStart + tid] ? 0.f : -1e9f;
    __syncthreads();
    if (kb + 1 < kbMax) {                       // prefetch next K/V block during compute
#pragma unroll
      for (int i = 0; i < 8; ++i) {
        int idx = tid + i * 256, row = idx >> 5, c2 = idx & 31;
        kReg[i] = KgBase[(kStart + 64 + row) * 32 + c2];
        vReg[i] = VgBase[(kStart + 64 + row) * 32 + c2];
      }
    }

    // S = Q*K^T : 4 key tiles x 2 d-steps
    v8f s[4] = {};
    float mAdd[4];
#pragma unroll
    for (int nt = 0; nt < 4; ++nt) {
#pragma unroll
      for (int ks = 0; ks < 2; ++ks) {
        v16bf bk = frag_from_lds(&Kfrag[(((nt * 2 + ks) * 32 + lane) << 3)]);
        s[nt] = __builtin_amdgcn_wmma_f32_16x16x32_bf16(false, aQ[ks], false, bk,
                                                        (short)0, s[nt], false, false);
      }
      mAdd[nt] = mAddS[nt * 16 + colL];
    }
    // causal + pad mask
#pragma unroll
    for (int nt = 0; nt < 4; ++nt) {
      int key = kStart + nt * 16 + colL;
#pragma unroll
      for (int r = 0; r < 8; ++r) {
        int qrow = q0 + rhalf + r;
        s[nt][r] = (key <= qrow) ? (s[nt][r] + mAdd[nt]) : -1e9f;
      }
    }
    // online softmax (per-row reductions across 16-lane halves)
    float mnew[8], pscale[8];
#pragma unroll
    for (int r = 0; r < 8; ++r) {
      float t4 = fmaxf(fmaxf(s[0][r], s[1][r]), fmaxf(s[2][r], s[3][r]));
#pragma unroll
      for (int off = 8; off; off >>= 1) t4 = fmaxf(t4, __shfl_xor(t4, off, 32));
      float mn = fmaxf(mrow[r], t4);
      pscale[r] = __expf(mrow[r] - mn);
      mrow[r] = mn; mnew[r] = mn;
    }
#pragma unroll
    for (int r = 0; r < 8; ++r) {
      float sum = 0.f;
#pragma unroll
      for (int nt = 0; nt < 4; ++nt) {
        float p = __expf(s[nt][r] - mnew[r]);
        s[nt][r] = p; sum += p;
      }
#pragma unroll
      for (int off = 8; off; off >>= 1) sum += __shfl_xor(sum, off, 32);
      lrow[r] = lrow[r] * pscale[r] + sum;
#pragma unroll
      for (int dt = 0; dt < 4; ++dt) o[dt][r] = o[dt][r] * pscale[r];
    }
    // P: C-layout registers -> A-layout LDS fragments (wave-private region)
#pragma unroll
    for (int nt = 0; nt < 4; ++nt) {
#pragma unroll
      for (int r = 0; r < 8; ++r) {
        int row = rhalf + r;
        int key = nt * 16 + colL;
        int ks = key >> 5, kkk = key & 31;
        int dlane = row + 16 * ((kkk >> 3) & 1);
        int elem = (kkk & 7) + 8 * ((kkk >> 4) & 1);
        PfragH[(((wave * 2 + ks) * 32 + dlane) << 4) + elem] =
            __builtin_bit_cast(u16, (__bf16)s[nt][r]);
      }
    }
    asm volatile("s_wait_dscnt 0" ::: "memory");   // same-wave LDS RAW
    v16bf aP[2];
#pragma unroll
    for (int ks = 0; ks < 2; ++ks)
      aP[ks] = frag_from_lds(&Pfrag[(((wave * 2 + ks) * 32 + lane) << 3)]);
    // O += P * V
#pragma unroll
    for (int dt = 0; dt < 4; ++dt) {
#pragma unroll
      for (int ks = 0; ks < 2; ++ks) {
        v16bf bv = frag_from_lds(&Vfrag[(((dt * 2 + ks) * 32 + lane) << 3)]);
        o[dt] = __builtin_amdgcn_wmma_f32_16x16x32_bf16(false, aP[ks], false, bv,
                                                        (short)0, o[dt], false, false);
      }
    }
  }
  // normalize + store context bf16 [B,T,C]
#pragma unroll
  for (int r = 0; r < 8; ++r) {
    float rl = lrow[r] > 0.f ? 1.0f / lrow[r] : 0.f;
    int t = q0 + rhalf + r;
    __bf16* cp = ctx + ((long long)(b * TT + t)) * CC + h * DH + colL;
#pragma unroll
    for (int dt = 0; dt < 4; ++dt)
      cp[dt * 16] = (__bf16)(o[dt][r] * rl);
  }
}

// ---------------------------------------------------------------- host
extern "C" void kernel_launch(void* const* d_in, const int* in_sizes, int n_in,
                              void* d_out, int out_size, void* d_ws, size_t ws_size,
                              hipStream_t stream) {
  const float* x     = (const float*)d_in[0];
  const int*   amask = (const int*)d_in[1];
  const float* Wqkv  = (const float*)d_in[2];
  const float* bqkv  = (const float*)d_in[3];
  const float* Wo    = (const float*)d_in[4];
  const float* bo    = (const float*)d_in[5];
  float* out = (float*)d_out;

  // workspace layout (bf16 buffers), ~92 MB total
  char* w = (char*)d_ws;
  u16* xbf   = (u16*)w;  w += (size_t)MROWS * CC * 2;
  u16* wqkvb = (u16*)w;  w += (size_t)CC * NQKV * 2;
  u16* wob   = (u16*)w;  w += (size_t)CC * CC * 2;
  u16* Qd    = (u16*)w;  w += (size_t)BSZ * HH * TT * DH * 2;
  u16* Kd    = (u16*)w;  w += (size_t)BSZ * HH * TT * DH * 2;
  u16* Vd    = (u16*)w;  w += (size_t)BSZ * HH * TT * DH * 2;
  u16* ctx   = (u16*)w;

  // 1) fp32 -> bf16 conversions
  cvt_f32_bf16_x4<<<1024, 256, 0, stream>>>((const float4*)x,    (ushort4*)xbf,   MROWS * CC / 4);
  cvt_f32_bf16_x4<<<1024, 256, 0, stream>>>((const float4*)Wqkv, (ushort4*)wqkvb, CC * NQKV / 4);
  cvt_f32_bf16_x4<<<512,  256, 0, stream>>>((const float4*)Wo,   (ushort4*)wob,   CC * CC / 4);

  // 2) QKV projection (M=8192, N=3072, K=1024)
  qkv_gemm_wmma<<<dim3(NQKV / 128, MROWS / 128), 256, 0, stream>>>(xbf, wqkvb, bqkv, Qd, Kd, Vd);

  // 3) causal flash attention per (b,h,128-row query block)
  attn_wmma<<<dim3(TT / 128, HH, BSZ), 256, 0, stream>>>(Qd, Kd, Vd, amask, ctx);

  // 4) output projection (M=8192, N=1024, K=1024) -> f32
  proj_gemm_wmma<<<dim3(CC / 128, MROWS / 128), 256, 0, stream>>>(ctx, wob, bo, out);
}
